// MemoryTGNN_44933947851145
// MI455X (gfx1250) — matrix operand order repaired and checked by reading, not compiled, not run on hardware
//
#include <hip/hip_runtime.h>

typedef __attribute__((ext_vector_type(16))) _Float16 v16h;
typedef __attribute__((ext_vector_type(8)))  _Float16 v8h;
typedef __attribute__((ext_vector_type(8)))  float    v8f;

#define N_NODES 50000
#define N_EDGES 800000
#define DIM_IN  128
#define DIM_H   64

// ---------------------------------------------------------------- helpers

__device__ __forceinline__ void wave_lds_fence() {
  // LDS ops from one wave complete in order; waiting DScnt==0 makes the
  // wave's own LDS stores visible to its own subsequent loads.
  asm volatile("s_wait_dscnt 0" ::: "memory");
}

__device__ __forceinline__ v8f wmma_f16(v16h a, v16h b, v8f c) {
  return __builtin_amdgcn_wmma_f32_16x16x32_f16(false, a, false, b, (short)0, c,
                                                false, false);
}

// Hardware transcendental path: v_exp_f32 / v_rcp_f32 are TRANS32 ops that
// co-execute with the WMMA pipe (ISA 5 sched notes), unlike the OCML
// polynomial expansions of expf/tanhf.
__device__ __forceinline__ float fast_rcp(float x) {
  return __builtin_amdgcn_rcpf(x);
}
__device__ __forceinline__ float fast_sigmoid(float x) {
  return fast_rcp(1.0f + __expf(-x));
}
__device__ __forceinline__ float fast_tanh(float x) {
  return 2.0f * fast_rcp(1.0f + __expf(-2.0f * x)) - 1.0f;
}

// Build a 16-half fragment from two 16-byte aligned runs (two b128 loads).
__device__ __forceinline__ v16h frag_2x16B(const _Float16* p0,
                                           const _Float16* p1) {
  union { v16h v; v8h h[2]; } u;
  u.h[0] = *(const v8h*)p0;
  u.h[1] = *(const v8h*)p1;
  return u.v;
}

// A fragment (16x32 f16) from one row-major f16 row.
// ISA 16-bit A layout: lanes 0-15 M=lane (K base 0), lanes 16-31 M=lane-16
// (K base +8); per lane the 16 halfs are K = kb..kb+7 and kb+16..kb+23,
// i.e. two contiguous 16-byte runs.
__device__ __forceinline__ v16h a_frag_row(const _Float16* __restrict__ row,
                                           int k0, int lane) {
  int kb = k0 + ((lane >> 4) << 3);
  return frag_2x16B(row + kb, row + kb + 16);
}

__device__ __forceinline__ v16h a_frag_global(const _Float16* __restrict__ p,
                                              int ld, int m0, int k0, int lane) {
  return a_frag_row(p + (size_t)(m0 + (lane & 15)) * ld, k0, lane);
}

// A fragment from an f16 LDS tile [16 rows, 64 halfs] (two ds_load_b128).
__device__ __forceinline__ v16h a_frag_lds16(const _Float16* __restrict__ sm,
                                             int k0, int lane) {
  const _Float16* row = sm + (lane & 15) * DIM_H;
  int kb = k0 + ((lane >> 4) << 3);
  return frag_2x16B(row + kb, row + kb + 16);
}

// B fragment from pre-packed fragment-major weights:
// packed[fidx*512 + lane*16 + j] = B[k, n],
//   n = (fidx % ntiles)*16 + (lane&15), k = (fidx/ntiles)*32 + (j>>3)*16 + (lane>>4)*8 + (j&7)
__device__ __forceinline__ v16h b_frag_packed(const _Float16* __restrict__ pk,
                                              int fidx, int lane) {
  const _Float16* p = pk + (((size_t)fidx * 32 + lane) << 4);
  return frag_2x16B(p, p + 8);
}

// monotone float <-> uint encoding for atomicMax on floats
__device__ __forceinline__ unsigned ord_from_float(float f) {
  unsigned b = __float_as_uint(f);
  return (b & 0x80000000u) ? ~b : (b | 0x80000000u);
}
__device__ __forceinline__ float float_from_ord(unsigned o) {
  return (o & 0x80000000u) ? __uint_as_float(o & 0x7fffffffu)
                           : __uint_as_float(~o);
}

// ---------------------------------------------------------------- converters

__global__ void cvt_f16_kernel(const float* __restrict__ s,
                               _Float16* __restrict__ d, long n) {
  long i = blockIdx.x * (long)blockDim.x + threadIdx.x;
  if (i < n) d[i] = (_Float16)s[i];
}

// Pack a [K, N] (or source-transposed [N, K]) f32 weight matrix into the
// fragment-major f16 layout consumed by b_frag_packed.
__global__ void pack_b_kernel(const float* __restrict__ src,
                              _Float16* __restrict__ dst, int K, int N,
                              int transpose) {
  int i = blockIdx.x * blockDim.x + threadIdx.x;
  if (i >= K * N) return;
  int ntiles = N >> 4;
  int f = i >> 9, r = i & 511;
  int lane = r >> 4, j = r & 15;
  int kc = f / ntiles, nt = f - kc * ntiles;
  int n = nt * 16 + (lane & 15);
  int k = kc * 32 + ((j >> 3) << 4) + ((lane >> 4) << 3) + (j & 7);
  float v = transpose ? src[(size_t)n * K + k] : src[(size_t)k * N + n];
  dst[i] = (_Float16)v;
}

// ------------------------------------------------- node encoder + node GRU

__global__ __launch_bounds__(128) void node_enc_gru_kernel(
    const _Float16* __restrict__ x16, const _Float16* __restrict__ Wenc,
    const float* __restrict__ b_enc, const float* __restrict__ lng,
    const float* __restrict__ lnb, const _Float16* __restrict__ WihT,
    const float* __restrict__ bih, const float* __restrict__ bhh,
    float* __restrict__ mem_out, float* __restrict__ hf,
    _Float16* __restrict__ h16) {
  __shared__ float sm_all[4 * 16 * 68];
  __shared__ __align__(16) _Float16 smh_all[4 * 16 * DIM_H];
  int lane = threadIdx.x & 31, wave = threadIdx.x >> 5;
  int tile = blockIdx.x * 4 + wave;
  if (tile >= N_NODES / 16) tile = N_NODES / 16 - 1;  // benign duplicate
  int m0 = tile * 16;
  float* sm = sm_all + wave * 16 * 68;
  _Float16* smh = smh_all + wave * 16 * DIM_H;
  int nl = lane & 15, hi = lane >> 4;

  // ---- encoder GEMM [16,128] x [128,64]   (B: ntiles=4, kchunks=4)
  v8f acc[4] = {};
#pragma unroll
  for (int kc = 0; kc < 4; ++kc) {
    v16h a = a_frag_global(x16, DIM_IN, m0, kc * 32, lane);
#pragma unroll
    for (int t = 0; t < 4; ++t)
      acc[t] = wmma_f16(a, b_frag_packed(Wenc, kc * 4 + t, lane), acc[t]);
  }
  // bias + leaky(0.01) into f32 LDS (column access for LN)
#pragma unroll
  for (int t = 0; t < 4; ++t) {
    int n = t * 16 + nl;
    float bn = b_enc[n];
#pragma unroll
    for (int r = 0; r < 8; ++r) {
      float v = acc[t][r] + bn;
      v = v > 0.f ? v : 0.01f * v;
      sm[(r + 8 * hi) * 68 + n] = v;
    }
  }
  wave_lds_fence();

  // ---- LayerNorm each row over 64 cols -> f16 LDS tile
  float g0 = lng[lane], g1 = lng[lane + 32], bb0 = lnb[lane], bb1 = lnb[lane + 32];
  for (int r = 0; r < 16; ++r) {
    float v0 = sm[r * 68 + lane], v1 = sm[r * 68 + 32 + lane];
    float s = v0 + v1, s2 = v0 * v0 + v1 * v1;
#pragma unroll
    for (int off = 16; off > 0; off >>= 1) {
      s += __shfl_down(s, off);
      s2 += __shfl_down(s2, off);
    }
    s = __shfl(s, 0); s2 = __shfl(s2, 0);
    float mu = s * (1.f / 64.f);
    float rs = rsqrtf(s2 * (1.f / 64.f) - mu * mu + 1e-5f);
    smh[r * DIM_H + lane]      = (_Float16)((v0 - mu) * rs * g0 + bb0);
    smh[r * DIM_H + 32 + lane] = (_Float16)((v1 - mu) * rs * g1 + bb1);
  }
  wave_lds_fence();

  // ---- GRU input GEMM [16,64] x [64,192]   (B: ntiles=12, kchunks=2)
  v16h ag0 = a_frag_lds16(smh, 0, lane);
  v16h ag1 = a_frag_lds16(smh, 32, lane);
  v8f g[12] = {};
#pragma unroll
  for (int t = 0; t < 12; ++t) {
    g[t] = wmma_f16(ag0, b_frag_packed(WihT, t, lane), g[t]);
    g[t] = wmma_f16(ag1, b_frag_packed(WihT, 12 + t, lane), g[t]);
  }
  // gates: hidden state is zero -> gh = bhh; memory = (1-z)*n
#pragma unroll
  for (int t = 0; t < 4; ++t) {
    int c = t * 16 + nl;
    float bir = bih[c], biz = bih[64 + c], bin = bih[128 + c];
    float bhr = bhh[c], bhz = bhh[64 + c], bhn = bhh[128 + c];
#pragma unroll
    for (int r = 0; r < 8; ++r) {
      float rg = fast_sigmoid(g[t][r] + bir + bhr);
      float zg = fast_sigmoid(g[t + 4][r] + biz + bhz);
      float ng = fast_tanh(g[t + 8][r] + bin + rg * bhn);
      float m_ = (1.f - zg) * ng;
      int row = m0 + r + 8 * hi;
      mem_out[row * DIM_H + c] = m_;
      hf[row * DIM_H + c]      = m_;
      h16[row * DIM_H + c]     = (_Float16)m_;
    }
  }
}

// ------------------------------------------------- edge projection + edge GRU

__global__ __launch_bounds__(128) void edge_proj_gru_kernel(
    const _Float16* __restrict__ h16, const int* __restrict__ src,
    const int* __restrict__ dst, const float* __restrict__ eattr,
    const _Float16* __restrict__ Wep, const float* __restrict__ b_ep,
    const _Float16* __restrict__ WeihT, const float* __restrict__ bih,
    const float* __restrict__ bhh, _Float16* __restrict__ ef16) {
  __shared__ __align__(16) _Float16 smh_all[4 * 16 * DIM_H];
  int lane = threadIdx.x & 31, wave = threadIdx.x >> 5;
  int tile = blockIdx.x * 4 + wave;
  if (tile >= N_EDGES / 16) tile = N_EDGES / 16 - 1;
  int e0 = tile * 16;
  _Float16* smh = smh_all + wave * 16 * DIM_H;
  int nl = lane & 15, hi = lane >> 4;

  int eL = e0 + nl;                 // each lane's A-fragment row
  const _Float16* rowS = h16 + (size_t)src[eL] * DIM_H;
  const _Float16* rowD = h16 + (size_t)dst[eL] * DIM_H;

  // ---- projection GEMM concat(h[src],h[dst]) [16,128] x [128,64]
  // K-chunks 0,1 read h[src] (K<64); chunks 2,3 read h[dst].
  v8f acc[4] = {};
#pragma unroll
  for (int kc = 0; kc < 4; ++kc) {
    const _Float16* row = (kc < 2) ? rowS : rowD;
    v16h a = a_frag_row(row, (kc & 1) * 32, lane);
#pragma unroll
    for (int t = 0; t < 4; ++t)
      acc[t] = wmma_f16(a, b_frag_packed(Wep, kc * 4 + t, lane), acc[t]);
  }
  // + b_ep + edge_attr -> f16 LDS (GRU input, no LN on edge path)
#pragma unroll
  for (int t = 0; t < 4; ++t) {
    int n = t * 16 + nl;
    float bn = b_ep[n];
#pragma unroll
    for (int r = 0; r < 8; ++r) {
      int er = e0 + r + 8 * hi;
      smh[(r + 8 * hi) * DIM_H + n] =
          (_Float16)(acc[t][r] + bn + eattr[(size_t)er * DIM_H + n]);
    }
  }
  wave_lds_fence();

  // ---- edge GRU GEMM [16,64] x [64,192]
  v16h ag0 = a_frag_lds16(smh, 0, lane);
  v16h ag1 = a_frag_lds16(smh, 32, lane);
  v8f g[12] = {};
#pragma unroll
  for (int t = 0; t < 12; ++t) {
    g[t] = wmma_f16(ag0, b_frag_packed(WeihT, t, lane), g[t]);
    g[t] = wmma_f16(ag1, b_frag_packed(WeihT, 12 + t, lane), g[t]);
  }
#pragma unroll
  for (int t = 0; t < 4; ++t) {
    int c = t * 16 + nl;
    float bir = bih[c], biz = bih[64 + c], bin = bih[128 + c];
    float bhr = bhh[c], bhz = bhh[64 + c], bhn = bhh[128 + c];
#pragma unroll
    for (int r = 0; r < 8; ++r) {
      float rg = fast_sigmoid(g[t][r] + bir + bhr);
      float zg = fast_sigmoid(g[t + 4][r] + biz + bhz);
      float ng = fast_tanh(g[t + 8][r] + bin + rg * bhn);
      int er = e0 + r + 8 * hi;
      ef16[(size_t)er * DIM_H + c] = (_Float16)((1.f - zg) * ng);
    }
  }
}

// ------------------------------------------------- GAT layer pieces

// hs = h @ W ; s_src[n,h] = <hs, a_src>, s_dst[n,h] = <hs, a_dst>
__global__ __launch_bounds__(128) void gat_node_kernel(
    const _Float16* __restrict__ h16, const _Float16* __restrict__ Wl,
    const float* __restrict__ asrc, const float* __restrict__ adst,
    float* __restrict__ hs, float* __restrict__ ssrc, float* __restrict__ sdst) {
  int lane = threadIdx.x & 31, wave = threadIdx.x >> 5;
  int tile = blockIdx.x * 4 + wave;
  if (tile >= N_NODES / 16) tile = N_NODES / 16 - 1;
  int m0 = tile * 16, nl = lane & 15, hi = lane >> 4;

  v8f acc[4] = {};
#pragma unroll
  for (int kc = 0; kc < 2; ++kc) {
    v16h a = a_frag_global(h16, DIM_H, m0, kc * 32, lane);
#pragma unroll
    for (int t = 0; t < 4; ++t)
      acc[t] = wmma_f16(a, b_frag_packed(Wl, kc * 4 + t, lane), acc[t]);
  }
#pragma unroll
  for (int t = 0; t < 4; ++t) {           // tile t == head t (16 dims)
    int n = t * 16 + nl;
    float av = asrc[n], dv = adst[n];
#pragma unroll
    for (int r = 0; r < 8; ++r) {
      int m = m0 + r + 8 * hi;
      float v = acc[t][r];
      hs[m * DIM_H + n] = v;
      float ps = v * av, pd = v * dv;
#pragma unroll
      for (int msk = 8; msk > 0; msk >>= 1) {
        ps += __shfl_xor(ps, msk);
        pd += __shfl_xor(pd, msk);
      }
      if (nl == 0) { ssrc[m * 4 + t] = ps; sdst[m * 4 + t] = pd; }
    }
  }
}

// e_alpha[e,h] = <ef @ We, a_edge>  (ee never materialized)
__global__ __launch_bounds__(128) void gat_edge_kernel(
    const _Float16* __restrict__ ef16, const _Float16* __restrict__ Wel,
    const float* __restrict__ aedge, float* __restrict__ ealpha) {
  int lane = threadIdx.x & 31, wave = threadIdx.x >> 5;
  int tile = blockIdx.x * 4 + wave;
  if (tile >= N_EDGES / 16) tile = N_EDGES / 16 - 1;
  int e0 = tile * 16, nl = lane & 15, hi = lane >> 4;

  v8f acc[4] = {};
#pragma unroll
  for (int kc = 0; kc < 2; ++kc) {
    v16h a = a_frag_global(ef16, DIM_H, e0, kc * 32, lane);
#pragma unroll
    for (int t = 0; t < 4; ++t)
      acc[t] = wmma_f16(a, b_frag_packed(Wel, kc * 4 + t, lane), acc[t]);
  }
#pragma unroll
  for (int t = 0; t < 4; ++t) {
    float av = aedge[t * 16 + nl];
#pragma unroll
    for (int r = 0; r < 8; ++r) {
      float ps = acc[t][r] * av;
#pragma unroll
      for (int msk = 8; msk > 0; msk >>= 1) ps += __shfl_xor(ps, msk);
      if (nl == 0) ealpha[(e0 + r + 8 * hi) * 4 + t] = ps;
    }
  }
}

// alpha = leaky(s_src[src]+s_dst[dst]+e_alpha, 0.2); segment max over dst
__global__ void alpha_max_kernel(const int* __restrict__ src,
                                 const int* __restrict__ dst,
                                 const float* __restrict__ ssrc,
                                 const float* __restrict__ sdst,
                                 float* __restrict__ ealpha,
                                 unsigned* __restrict__ nmax) {
  int i = blockIdx.x * blockDim.x + threadIdx.x;
  if (i >= N_EDGES * 4) return;
  int e = i >> 2, hh = i & 3;
  float a = ssrc[src[e] * 4 + hh] + sdst[dst[e] * 4 + hh] + ealpha[i];
  a = a > 0.f ? a : 0.2f * a;
  ealpha[i] = a;
  atomicMax(&nmax[dst[e] * 4 + hh], ord_from_float(a));
}

// w = exp(alpha - max[dst]); segment sum over dst
__global__ void exp_sum_kernel(const int* __restrict__ dst,
                               const unsigned* __restrict__ nmax,
                               float* __restrict__ ealpha,
                               float* __restrict__ nsum) {
  int i = blockIdx.x * blockDim.x + threadIdx.x;
  if (i >= N_EDGES * 4) return;
  int e = i >> 2, hh = i & 3;
  float w = __expf(ealpha[i] - float_from_ord(nmax[dst[e] * 4 + hh]));
  ealpha[i] = w;
  atomicAdd(&nsum[dst[e] * 4 + hh], w);
}

// agg[dst] += hs[src] * w
__global__ void aggregate_kernel(const int* __restrict__ src,
                                 const int* __restrict__ dst,
                                 const float* __restrict__ hs,
                                 const float* __restrict__ ealpha,
                                 float* __restrict__ agg) {
  long i = blockIdx.x * (long)blockDim.x + threadIdx.x;
  if (i >= (long)N_EDGES * DIM_H) return;
  int e = (int)(i >> 6), c = (int)(i & 63), hh = c >> 4;
  float w = ealpha[e * 4 + hh];
  atomicAdd(&agg[dst[e] * DIM_H + c], hs[src[e] * DIM_H + c] * w);
}

// h = LN(h + relu(agg/nsum + bias)) ; refresh f16 copy. One wave per node.
__global__ __launch_bounds__(128) void gat_finish_kernel(
    const float* __restrict__ agg, const float* __restrict__ nsum,
    const float* __restrict__ bias, const float* __restrict__ lng,
    const float* __restrict__ lnb, float* __restrict__ hf,
    _Float16* __restrict__ h16) {
  int lane = threadIdx.x & 31, wave = threadIdx.x >> 5;
  int node = blockIdx.x * 4 + wave;
  if (node >= N_NODES) node = N_NODES - 1;
  int c0 = lane, c1 = lane + 32;
  float r0 = fast_rcp(nsum[node * 4 + (lane >> 4)] + 1e-16f);
  float r1 = fast_rcp(nsum[node * 4 + 2 + (lane >> 4)] + 1e-16f);
  float v0 = agg[node * DIM_H + c0] * r0 + bias[c0];
  float v1 = agg[node * DIM_H + c1] * r1 + bias[c1];
  float h0 = hf[node * DIM_H + c0] + fmaxf(v0, 0.f);
  float h1 = hf[node * DIM_H + c1] + fmaxf(v1, 0.f);
  float s = h0 + h1, s2 = h0 * h0 + h1 * h1;
#pragma unroll
  for (int off = 16; off > 0; off >>= 1) {
    s += __shfl_down(s, off);
    s2 += __shfl_down(s2, off);
  }
  s = __shfl(s, 0); s2 = __shfl(s2, 0);
  float mu = s * (1.f / 64.f);
  float rs = rsqrtf(s2 * (1.f / 64.f) - mu * mu + 1e-5f);
  float o0 = (h0 - mu) * rs * lng[c0] + lnb[c0];
  float o1 = (h1 - mu) * rs * lng[c1] + lnb[c1];
  hf[node * DIM_H + c0] = o0;  h16[node * DIM_H + c0] = (_Float16)o0;
  hf[node * DIM_H + c1] = o1;  h16[node * DIM_H + c1] = (_Float16)o1;
}

// ------------------------------------------------- final projection + LN + L2

__global__ __launch_bounds__(128) void final_kernel(
    const _Float16* __restrict__ h16, const _Float16* __restrict__ Wnp,
    const float* __restrict__ b_np, const float* __restrict__ lng,
    const float* __restrict__ lnb, float* __restrict__ z_out) {
  __shared__ float sm_all[4 * 16 * 68];
  int lane = threadIdx.x & 31, wave = threadIdx.x >> 5;
  int tile = blockIdx.x * 4 + wave;
  if (tile >= N_NODES / 16) tile = N_NODES / 16 - 1;
  int m0 = tile * 16, nl = lane & 15, hi = lane >> 4;
  float* sm = sm_all + wave * 16 * 68;

  v8f acc[4] = {};
#pragma unroll
  for (int kc = 0; kc < 2; ++kc) {
    v16h a = a_frag_global(h16, DIM_H, m0, kc * 32, lane);
#pragma unroll
    for (int t = 0; t < 4; ++t)
      acc[t] = wmma_f16(a, b_frag_packed(Wnp, kc * 4 + t, lane), acc[t]);
  }
#pragma unroll
  for (int t = 0; t < 4; ++t) {
    int n = t * 16 + nl;
    float bn = b_np[n];
#pragma unroll
    for (int r = 0; r < 8; ++r) {
      float v = acc[t][r] + bn;
      v = v > 0.f ? v : 0.01f * v;
      sm[(r + 8 * hi) * 68 + n] = v;
    }
  }
  wave_lds_fence();

  float g0 = lng[lane], g1 = lng[lane + 32], bb0 = lnb[lane], bb1 = lnb[lane + 32];
  for (int r = 0; r < 16; ++r) {
    float v0 = sm[r * 68 + lane], v1 = sm[r * 68 + 32 + lane];
    float s = v0 + v1, s2 = v0 * v0 + v1 * v1;
#pragma unroll
    for (int off = 16; off > 0; off >>= 1) {
      s += __shfl_down(s, off);
      s2 += __shfl_down(s2, off);
    }
    s = __shfl(s, 0); s2 = __shfl(s2, 0);
    float mu = s * (1.f / 64.f);
    float rs = rsqrtf(s2 * (1.f / 64.f) - mu * mu + 1e-5f);
    float n0 = (v0 - mu) * rs * g0 + bb0;
    float n1 = (v1 - mu) * rs * g1 + bb1;
    float q = n0 * n0 + n1 * n1;
#pragma unroll
    for (int off = 16; off > 0; off >>= 1) q += __shfl_down(q, off);
    q = __shfl(q, 0);
    float inv = fast_rcp(fmaxf(sqrtf(q), 1e-12f));
    int row = m0 + r;
    z_out[row * DIM_H + lane]      = n0 * inv;
    z_out[row * DIM_H + 32 + lane] = n1 * inv;
  }
}

// ---------------------------------------------------------------- launch

extern "C" void kernel_launch(void* const* d_in, const int* in_sizes, int n_in,
                              void* d_out, int out_size, void* d_ws,
                              size_t ws_size, hipStream_t stream) {
  (void)in_sizes; (void)n_in; (void)out_size; (void)ws_size;
  const float* x         = (const float*)d_in[0];
  const int*   eidx      = (const int*)d_in[1];
  const float* eattr     = (const float*)d_in[2];
  const float* W_enc     = (const float*)d_in[3];
  const float* b_enc     = (const float*)d_in[4];
  const float* ln_enc_g  = (const float*)d_in[5];
  const float* ln_enc_b  = (const float*)d_in[6];
  const float* gru_n_Wih = (const float*)d_in[7];
  const float* gru_n_bih = (const float*)d_in[9];
  const float* gru_n_bhh = (const float*)d_in[10];
  const float* W_ep      = (const float*)d_in[11];
  const float* b_ep      = (const float*)d_in[12];
  const float* gru_e_Wih = (const float*)d_in[13];
  const float* gru_e_bih = (const float*)d_in[15];
  const float* gru_e_bhh = (const float*)d_in[16];
  const float* gat_W     = (const float*)d_in[17];
  const float* gat_We    = (const float*)d_in[18];
  const float* gat_asrc  = (const float*)d_in[19];
  const float* gat_adst  = (const float*)d_in[20];
  const float* gat_aedge = (const float*)d_in[21];
  const float* gat_bias  = (const float*)d_in[22];
  const float* ln_g      = (const float*)d_in[23];
  const float* ln_b      = (const float*)d_in[24];
  const float* W_np      = (const float*)d_in[25];
  const float* b_np      = (const float*)d_in[26];
  const float* ln_np_g   = (const float*)d_in[27];
  const float* ln_np_b   = (const float*)d_in[28];

  const int* srcI = eidx;
  const int* dstI = eidx + N_EDGES;
  float* z_out   = (float*)d_out;
  float* mem_out = (float*)d_out + (size_t)N_NODES * DIM_H;

  // workspace carve-out (all 256-byte aligned)
  char* wp = (char*)d_ws;
  auto carve = [&](size_t bytes) {
    void* p = (void*)wp;
    wp += (bytes + 255) & ~(size_t)255;
    return p;
  };
  _Float16* x16     = (_Float16*)carve((size_t)N_NODES * DIM_IN * 2);
  _Float16* h16     = (_Float16*)carve((size_t)N_NODES * DIM_H * 2);
  _Float16* ef16    = (_Float16*)carve((size_t)N_EDGES * DIM_H * 2);
  float*    hf      = (float*)carve((size_t)N_NODES * DIM_H * 4);
  float*    hs      = (float*)carve((size_t)N_NODES * DIM_H * 4);
  float*    agg     = (float*)carve((size_t)N_NODES * DIM_H * 4);
  float*    ssrc    = (float*)carve((size_t)N_NODES * 4 * 4);
  float*    sdst    = (float*)carve((size_t)N_NODES * 4 * 4);
  unsigned* nmax    = (unsigned*)carve((size_t)N_NODES * 4 * 4);
  float*    nsum    = (float*)carve((size_t)N_NODES * 4 * 4);
  float*    ealpha  = (float*)carve((size_t)N_EDGES * 4 * 4);
  _Float16* Wenc16  = (_Float16*)carve((size_t)DIM_IN * DIM_H * 2);
  _Float16* WihT16  = (_Float16*)carve((size_t)DIM_H * 192 * 2);
  _Float16* Wep16   = (_Float16*)carve((size_t)DIM_IN * DIM_H * 2);
  _Float16* WeihT16 = (_Float16*)carve((size_t)DIM_H * 192 * 2);
  _Float16* gatW16  = (_Float16*)carve((size_t)3 * DIM_H * DIM_H * 2);
  _Float16* gatWe16 = (_Float16*)carve((size_t)3 * DIM_H * DIM_H * 2);
  _Float16* Wnp16   = (_Float16*)carve((size_t)DIM_H * DIM_H * 2);

  auto gridOf = [](long n) { return dim3((unsigned)((n + 255) / 256)); };

  // f16 / fragment-packing pre-pass
  cvt_f16_kernel<<<gridOf((long)N_NODES * DIM_IN), 256, 0, stream>>>(
      x, x16, (long)N_NODES * DIM_IN);
  pack_b_kernel<<<gridOf(DIM_IN * DIM_H), 256, 0, stream>>>(
      W_enc, Wenc16, DIM_IN, DIM_H, 0);
  pack_b_kernel<<<gridOf(DIM_H * 192), 256, 0, stream>>>(
      gru_n_Wih, WihT16, DIM_H, 192, 1);   // src [192,64] -> B [64,192]
  pack_b_kernel<<<gridOf(DIM_IN * DIM_H), 256, 0, stream>>>(
      W_ep, Wep16, DIM_IN, DIM_H, 0);
  pack_b_kernel<<<gridOf(DIM_H * 192), 256, 0, stream>>>(
      gru_e_Wih, WeihT16, DIM_H, 192, 1);
  for (int l = 0; l < 3; ++l) {
    pack_b_kernel<<<gridOf(DIM_H * DIM_H), 256, 0, stream>>>(
        gat_W + (size_t)l * 4096, gatW16 + (size_t)l * 4096, DIM_H, DIM_H, 0);
    pack_b_kernel<<<gridOf(DIM_H * DIM_H), 256, 0, stream>>>(
        gat_We + (size_t)l * 4096, gatWe16 + (size_t)l * 4096, DIM_H, DIM_H, 0);
  }
  pack_b_kernel<<<gridOf(DIM_H * DIM_H), 256, 0, stream>>>(
      W_np, Wnp16, DIM_H, DIM_H, 0);

  // node encoder + node GRU (memory)
  const int NB_N = (N_NODES / 16 + 3) / 4;   // 782
  node_enc_gru_kernel<<<NB_N, 128, 0, stream>>>(x16, Wenc16, b_enc, ln_enc_g,
      ln_enc_b, WihT16, gru_n_bih, gru_n_bhh, mem_out, hf, h16);

  // edge projection + edge GRU
  const int NB_E = N_EDGES / 16 / 4;         // 12500
  edge_proj_gru_kernel<<<NB_E, 128, 0, stream>>>(h16, srcI, dstI, eattr, Wep16,
      b_ep, WeihT16, gru_e_bih, gru_e_bhh, ef16);

  // 3 GAT layers
  for (int l = 0; l < 3; ++l) {
    hipMemsetAsync(nmax, 0, (size_t)N_NODES * 4 * 4, stream);
    hipMemsetAsync(nsum, 0, (size_t)N_NODES * 4 * 4, stream);
    hipMemsetAsync(agg, 0, (size_t)N_NODES * DIM_H * 4, stream);

    gat_node_kernel<<<NB_N, 128, 0, stream>>>(h16, gatW16 + (size_t)l * 4096,
        gat_asrc + l * 64, gat_adst + l * 64, hs, ssrc, sdst);
    gat_edge_kernel<<<NB_E, 128, 0, stream>>>(ef16, gatWe16 + (size_t)l * 4096,
        gat_aedge + l * 64, ealpha);
    alpha_max_kernel<<<gridOf((long)N_EDGES * 4), 256, 0, stream>>>(
        srcI, dstI, ssrc, sdst, ealpha, nmax);
    exp_sum_kernel<<<gridOf((long)N_EDGES * 4), 256, 0, stream>>>(
        dstI, nmax, ealpha, nsum);
    aggregate_kernel<<<gridOf((long)N_EDGES * DIM_H), 256, 0, stream>>>(
        srcI, dstI, hs, ealpha, agg);
    gat_finish_kernel<<<(N_NODES + 3) / 4, 128, 0, stream>>>(agg, nsum,
        gat_bias + l * 64, ln_g + l * 64, ln_b + l * 64, hf, h16);
  }

  // final projection + LayerNorm + L2 normalize -> z
  final_kernel<<<NB_N, 128, 0, stream>>>(h16, Wnp16, b_np, ln_np_g, ln_np_b, z_out);
}
